// HOIMLoss_57741540327610
// MI455X (gfx1250) — matrix operand order repaired and checked by reading, not compiled, not run on hardware
//
#include <hip/hip_runtime.h>

#define NUM_PIDS 5532
#define NUM_CQ   5000
#define NUM_BG   5000
#define NF       256
#define BATCH    4096
#define NTAB     (NUM_BG + NUM_PIDS + NUM_CQ)   /* 15532 */
#define TPAD     15552                          /* padded table rows: 972 tiles */
#define NPAIR    486                            /* 486 tile-pairs of 32 columns */
#define NSEG     4
#define SEGPAIRS ((NPAIR + NSEG - 1) / NSEG)    /* 122 */

typedef __attribute__((ext_vector_type(16))) _Float16 v16h;
typedef __attribute__((ext_vector_type(8)))  _Float16 v8h;
typedef __attribute__((ext_vector_type(8)))  float    v8f;
typedef __attribute__((ext_vector_type(4)))  int      v4i;

// ---------------------------------------------------------------------------
// CDNA5 async global->LDS copy (ASYNCcnt-tracked) + wait, with asm fallback.
// Builtin signature (probe-confirmed): (v4i AS1*, v4i AS3*, imm off, imm cpol).
// ---------------------------------------------------------------------------
__device__ __forceinline__ void async_copy16(const _Float16* g, _Float16* l) {
#if __has_builtin(__builtin_amdgcn_global_load_async_to_lds_b128)
  __builtin_amdgcn_global_load_async_to_lds_b128(
      (__attribute__((address_space(1))) v4i*)g,
      (__attribute__((address_space(3))) v4i*)l, 0, 0);
#else
  __attribute__((address_space(3))) _Float16* l3 =
      (__attribute__((address_space(3))) _Float16*)l;
  unsigned loff = (unsigned)(size_t)l3;
  asm volatile("global_load_async_to_lds_b128 %0, %1, off"
               :: "v"(loff), "v"(g) : "memory");
#endif
}

__device__ __forceinline__ void wait_async0() {
#if __has_builtin(__builtin_amdgcn_s_wait_asynccnt)
  __builtin_amdgcn_s_wait_asynccnt(0);
#else
  asm volatile("s_wait_asynccnt 0" ::: "memory");
#endif
}

// ---------------------------------------------------------------------------
// Phase 0: build f16 copies of X and the concatenated table (zero-padded).
// ---------------------------------------------------------------------------
__global__ void hoim_convert_kernel(const float* __restrict__ inp,
                                    const float* __restrict__ lut,
                                    const float* __restrict__ cq,
                                    const float* __restrict__ cqb,
                                    _Float16* __restrict__ Tf16,
                                    _Float16* __restrict__ Xf16) {
  const int total_t = TPAD * NF;
  const int total_x = BATCH * NF;
  for (int i = blockIdx.x * blockDim.x + threadIdx.x; i < total_t + total_x;
       i += gridDim.x * blockDim.x) {
    if (i < total_t) {
      int row = i >> 8;
      int k   = i & (NF - 1);
      float v = 0.0f;
      if (row < NUM_BG)                 v = cqb[row * NF + k];
      else if (row < NUM_BG + NUM_PIDS) v = lut[(row - NUM_BG) * NF + k];
      else if (row < NTAB)              v = cq[(row - NUM_BG - NUM_PIDS) * NF + k];
      Tf16[i] = (_Float16)v;
    } else {
      int j = i - total_t;
      Xf16[j] = (_Float16)inp[j];
    }
  }
}

__device__ __forceinline__ v16h ldb16(const _Float16* p) {
  v8h lo = *(const v8h*)(p);
  v8h hb = *(const v8h*)(p + 16);
  return __builtin_shufflevector(lo, hb, 0,1,2,3,4,5,6,7,8,9,10,11,12,13,14,15);
}

__device__ __forceinline__ void accum_stats(const v8f& acc, int col,
                                            const int (&labv)[8],
                                            float (&s_bg)[8], float (&s_nb)[8],
                                            float (&vlab)[8]) {
  if (col < NTAB) {
    if (col < NUM_BG) {
#pragma unroll
      for (int r = 0; r < 8; ++r)
        s_bg[r] += __expf(fmaf(30.0f, acc[r], -30.0f));
    } else {
      const int nbcol = col - NUM_BG;
#pragma unroll
      for (int r = 0; r < 8; ++r) {
        float ls = fmaf(30.0f, acc[r], -30.0f);   // shifted logit (<= 0)
        s_nb[r] += __expf(ls);
        if (nbcol == labv[r]) vlab[r] = ls;
      }
    }
  }
}

// ---------------------------------------------------------------------------
// Phase 1: 4 waves x 16 rows per block, one column segment per blockIdx.y.
// 32-column tile-pairs staged to double-buffered LDS via async copy; two
// independent WMMA accumulator chains per wave for in-wave ILP.
// Logits are bounded (|dot|<=1 -> |logit|<=30), so fixed-shift exp sums are
// numerically safe: no online max needed, partials are associative.
// ---------------------------------------------------------------------------
__global__ __launch_bounds__(128) void hoim_gemm_kernel(
    const _Float16* __restrict__ Xf16,
    const _Float16* __restrict__ Tf16,
    const int*      __restrict__ roi_label,
    float*          __restrict__ sbg_p,    // [NSEG][BATCH]
    float*          __restrict__ snb_p,    // [NSEG][BATCH]
    float*          __restrict__ vlab_p)   // [NSEG][BATCH]
{
  __shared__ _Float16 lds_b[2][32 * NF];   // 2 x 16 KB double buffer

  const int tid     = threadIdx.x;
  const int lane    = tid & 31;
  const int wave    = tid >> 5;
  const int lhalf   = lane & 15;
  const int hi      = (lane >> 4) & 1;
  const int rowbase = blockIdx.x * 64 + wave * 16;
  const int seg     = blockIdx.y;
  const int p_lo    = seg * SEGPAIRS;
  const int p_hi    = (p_lo + SEGPAIRS < NPAIR) ? (p_lo + SEGPAIRS) : NPAIR;

  // ---- A fragments for all 8 K-chunks (CDNA5 16-bit A layout):
  // lane<16: row=lane, K = k0+{0..7} | k0+{16..23}; lane>=16: +8 / +24 octets
  v16h a[8];
  {
    const _Float16* xrow = Xf16 + (size_t)(rowbase + lhalf) * NF + hi * 8;
#pragma unroll
    for (int c = 0; c < 8; ++c) {
      v8h lo = *(const v8h*)(xrow + c * 32);
      v8h hb = *(const v8h*)(xrow + c * 32 + 16);
      a[c] = __builtin_shufflevector(lo, hb, 0,1,2,3,4,5,6,7,8,9,10,11,12,13,14,15);
    }
  }

  int labv[8];
#pragma unroll
  for (int r = 0; r < 8; ++r) labv[r] = roi_label[rowbase + hi * 8 + r];

  float s_bg[8], s_nb[8], vlab[8];
#pragma unroll
  for (int r = 0; r < 8; ++r) { s_bg[r] = 0.0f; s_nb[r] = 0.0f; vlab[r] = -1e30f; }

  // stage one 32-row B tile-pair (16 KB) into LDS: 128 thr x 8 x 16 B
#define STAGE_PAIR(P, BUF)                                            \
  do {                                                                \
    const _Float16* gsrc = Tf16 + (size_t)(P) * 32 * NF;              \
    _Float16* ldst = &lds_b[(BUF)][0];                                \
    _Pragma("unroll")                                                 \
    for (int i_ = 0; i_ < 8; ++i_) {                                  \
      int off_ = (tid + i_ * 128) * 8;                                \
      async_copy16(gsrc + off_, ldst + off_);                         \
    }                                                                 \
  } while (0)

  STAGE_PAIR(p_lo, p_lo & 1);

  for (int p = p_lo; p < p_hi; ++p) {
    wait_async0();        // own async copies for pair p have landed
    __syncthreads();      // everyone's copies landed; prior reads done
    if (p + 1 < p_hi) STAGE_PAIR(p + 1, (p + 1) & 1);   // prefetch next

    // B layout: lane<16: col=lane, K=k0+{0..7}|{16..23} ; lane>=16: col=lane-16, K=k0+{8..15}|{24..31}
    const _Float16* base  = &lds_b[p & 1][0];
    const _Float16* bsrc0 = base + (size_t)lhalf * NF + hi * 8;
    const _Float16* bsrc1 = base + (size_t)(16 + lhalf) * NF + hi * 8;

    v8f acc0 = {}, acc1 = {};
#pragma unroll
    for (int c = 0; c < 8; ++c) {
      v16h b0 = ldb16(bsrc0 + c * 32);
      v16h b1 = ldb16(bsrc1 + c * 32);
      acc0 = __builtin_amdgcn_wmma_f32_16x16x32_f16(false, a[c], false, b0,
                                                    (short)0, acc0, false, false);
      acc1 = __builtin_amdgcn_wmma_f32_16x16x32_f16(false, a[c], false, b1,
                                                    (short)0, acc1, false, false);
    }

    // D layout: VGPR r -> row (r + 8*hi), column = lhalf within each tile.
    const int cb = p * 32 + lhalf;
    accum_stats(acc0, cb,      labv, s_bg, s_nb, vlab);
    accum_stats(acc1, cb + 16, labv, s_bg, s_nb, vlab);
  }
#undef STAGE_PAIR

  // merge the 16 lanes holding each row's column subsets (fixed order)
#pragma unroll
  for (int mask = 1; mask <= 8; mask <<= 1) {
#pragma unroll
    for (int r = 0; r < 8; ++r) {
      s_bg[r] += __shfl_xor(s_bg[r], mask);
      s_nb[r] += __shfl_xor(s_nb[r], mask);
      vlab[r]  = fmaxf(vlab[r], __shfl_xor(vlab[r], mask));
    }
  }

  if (lhalf == 0) {  // lane 0 -> rows 0..7, lane 16 -> rows 8..15
#pragma unroll
    for (int r = 0; r < 8; ++r) {
      const int idx = seg * BATCH + rowbase + hi * 8 + r;
      sbg_p[idx]  = s_bg[r];
      snb_p[idx]  = s_nb[r];
      vlab_p[idx] = vlab[r];
    }
  }
}

// ---------------------------------------------------------------------------
// Phase 2: per-row combine of segment partials -> cls_score + loss terms.
// ---------------------------------------------------------------------------
__global__ __launch_bounds__(256) void hoim_rowfin_kernel(
    const float* __restrict__ sbg_p,
    const float* __restrict__ snb_p,
    const float* __restrict__ vlab_p,
    const int*   __restrict__ roi_label,
    float* __restrict__ cls_out,       // d_out[0..8191]
    float* __restrict__ det_terms,
    float* __restrict__ oim_terms,
    float* __restrict__ valid_flags)
{
  const int row = blockIdx.x * blockDim.x + threadIdx.x;
  if (row >= BATCH) return;

  float sb = 0.0f, sn = 0.0f, vl = -1e30f;
#pragma unroll
  for (int s = 0; s < NSEG; ++s) {
    sb += sbg_p[s * BATCH + row];
    sn += snb_p[s * BATCH + row];
    vl  = fmaxf(vl, vlab_p[s * BATCH + row]);
  }

  const int lb = roi_label[row];
  float inv = 1.0f / (sb + sn);
  float c0  = sb * inv;
  float c1  = sn * inv;
  cls_out[row * 2 + 0] = c0;
  cls_out[row * 2 + 1] = c1;

  float rl = 0.0f;
  if (lb >= 0) {                       // labeled: focal log-softmax at label
    float logp = vl - __logf(sn);      // both shifted by the same -30
    float p    = __expf(logp);
    float om1  = 1.0f - p;
    rl = -(0.25f * om1 * om1 * logp) * (c1 * c1);
  }
  bool valid = (lb >= -1);
  oim_terms[row]   = valid ? rl : 0.0f;
  valid_flags[row] = valid ? 1.0f : 0.0f;

  float cs = (lb == -2) ? c0 : c1;     // det_label = clip(lb+2,0,1)
  float o  = 1.0f - cs;
  det_terms[row] = -(0.25f * o * o * __logf(cs));
}

// ---------------------------------------------------------------------------
// Phase 3: deterministic single-block reduction for the two scalar losses.
// ---------------------------------------------------------------------------
__global__ __launch_bounds__(256) void hoim_finalize_kernel(
    const float* __restrict__ det_terms,
    const float* __restrict__ oim_terms,
    const float* __restrict__ valid_flags,
    float* __restrict__ out2)   // d_out + 8192 : {loss_det, loss_oim}
{
  __shared__ float sd[256], so[256], sv[256];
  const int t = threadIdx.x;
  float a = 0.0f, b = 0.0f, c = 0.0f;
  for (int i = t; i < BATCH; i += 256) {
    a += det_terms[i];
    b += oim_terms[i];
    c += valid_flags[i];
  }
  sd[t] = a; so[t] = b; sv[t] = c;
  __syncthreads();
  for (int s = 128; s > 0; s >>= 1) {
    if (t < s) { sd[t] += sd[t + s]; so[t] += so[t + s]; sv[t] += sv[t + s]; }
    __syncthreads();
  }
  if (t == 0) {
    out2[0] = sd[0] / (float)BATCH;
    out2[1] = so[0] / fmaxf(sv[0], 1.0f);
  }
}

// ---------------------------------------------------------------------------
extern "C" void kernel_launch(void* const* d_in, const int* in_sizes, int n_in,
                              void* d_out, int out_size, void* d_ws, size_t ws_size,
                              hipStream_t stream) {
  const float* inputs = (const float*)d_in[0];
  const int*   roi    = (const int*)  d_in[1];
  const float* lut    = (const float*)d_in[2];
  const float* cq     = (const float*)d_in[3];
  const float* cqb    = (const float*)d_in[4];
  float* out = (float*)d_out;

  char* ws = (char*)d_ws;
  size_t o = 0;
  _Float16* Tf16 = (_Float16*)(ws + o); o += (size_t)TPAD * NF * 2;
  _Float16* Xf16 = (_Float16*)(ws + o); o += (size_t)BATCH * NF * 2;
  float* sbg_p   = (float*)(ws + o);    o += (size_t)NSEG * BATCH * 4;
  float* snb_p   = (float*)(ws + o);    o += (size_t)NSEG * BATCH * 4;
  float* vlab_p  = (float*)(ws + o);    o += (size_t)NSEG * BATCH * 4;
  float* det_t   = (float*)(ws + o);    o += (size_t)BATCH * 4;
  float* oim_t   = (float*)(ws + o);    o += (size_t)BATCH * 4;
  float* valid_f = (float*)(ws + o);    o += (size_t)BATCH * 4;

  hipLaunchKernelGGL(hoim_convert_kernel, dim3(1024), dim3(256), 0, stream,
                     inputs, lut, cq, cqb, Tf16, Xf16);
  hipLaunchKernelGGL(hoim_gemm_kernel, dim3(BATCH / 64, NSEG), dim3(128), 0, stream,
                     Xf16, Tf16, roi, sbg_p, snb_p, vlab_p);
  hipLaunchKernelGGL(hoim_rowfin_kernel, dim3(BATCH / 256), dim3(256), 0, stream,
                     sbg_p, snb_p, vlab_p, roi, out, det_t, oim_t, valid_f);
  hipLaunchKernelGGL(hoim_finalize_kernel, dim3(1), dim3(256), 0, stream,
                     det_t, oim_t, valid_f, out + (size_t)BATCH * 2);
}
